// GPR_1932735283957
// MI455X (gfx1250) — compile-verified, matrix operand order checked by reading, not run on hardware
//
#include <hip/hip_runtime.h>
#include <hip/hip_bf16.h>

// ---------------------------------------------------------------------------
// GPRGNN forward on MI455X (gfx1250, wave32, WMMA).
//  - GEMMs: v_wmma_f32_16x16x32_bf16, bf16 operands / fp32 accumulate.
//  - Weights pre-packed once into bf16 fragment-major layout.
//  - Layer/output GEMMs stage packed B into LDS via async global->LDS copies
//    (ASYNCcnt), then read fragments with ds_load_b128: L0/L1 stays free for
//    the streamed A operand, which is the bound resource at 23.3 TB/s.
// ---------------------------------------------------------------------------

typedef __attribute__((ext_vector_type(16))) __bf16 v16bf;
typedef __attribute__((ext_vector_type(8)))  __bf16 v8bf;
typedef __attribute__((ext_vector_type(4)))  __bf16 v4bf;
typedef __attribute__((ext_vector_type(8)))  float  v8f;
typedef int v4i __attribute__((vector_size(16)));   // matches async-LDS builtin

#define AS1 __attribute__((address_space(1)))
#define AS3 __attribute__((address_space(3)))

__device__ __forceinline__ __bf16 f2bf(float f) {
    union { float f; unsigned u; } a; a.f = f;
    unsigned r = a.u + 0x7FFFu + ((a.u >> 16) & 1u);   // RNE
    unsigned short h = (unsigned short)(r >> 16);
    __bf16 b;
    __builtin_memcpy(&b, &h, sizeof(b));
    return b;
}

// Fragment element j (0..15) of a 16-bit 16x32 A/B fragment holds
// K = kgrp + j + (j>=8 ? 8 : 0), kgrp = (lane>=16)*8  -> two contiguous runs.

// Pack B[K x Nn] (f32 row-major) into fragment-major bf16:
// Bpack[((kc32*NT + t)*32 + lane)*16 + j]
__global__ void gpr_pack_b(const float* __restrict__ W, __bf16* __restrict__ Bpack,
                           int Nn, int NT, int nfrag) {
    const int tid  = blockIdx.x * blockDim.x + threadIdx.x;
    const int f    = tid >> 5;
    const int lane = tid & 31;
    if (f >= nfrag) return;
    const int kc32 = f / NT;
    const int t    = f % NT;
    const int kgrp = (lane >> 4) << 3;
    const int col  = t * 16 + (lane & 15);
    __bf16* outp = Bpack + ((size_t)f * 32 + lane) * 16;
#pragma unroll
    for (int j = 0; j < 16; ++j) {
        const int k = kc32 * 32 + kgrp + j + ((j >= 8) ? 8 : 0);
        outp[j] = f2bf(W[(size_t)k * Nn + col]);
    }
}

// Row-major f32 -> bf16, 4 elements per thread (n % 4 == 0).
__global__ void gpr_f32_to_bf16(const float* __restrict__ in,
                                __bf16* __restrict__ out, size_t n4) {
    size_t i = (size_t)blockIdx.x * blockDim.x + threadIdx.x;
    if (i >= n4) return;
    const float4 f = ((const float4*)in)[i];
    v4bf b;
    b[0] = f2bf(f.x); b[1] = f2bf(f.y); b[2] = f2bf(f.z); b[3] = f2bf(f.w);
    ((v4bf*)out)[i] = b;
}

// C[M x Nn] = A[M x K](bf16) * Bpack(+bias), K = KC*32, Nn = NT*16.
// Optional outputs: Cf (f32), Cbf (bf16, feeds next GEMM's A), H2 = scale*res.
// One wave per 16-row strip; M%16==0.
template<int NT, int KC, bool USE_LDS>
__global__ void gpr_gemm_wmma(const __bf16* __restrict__ A,
                              const __bf16* __restrict__ Bpack,
                              const float* __restrict__ bias,
                              float* __restrict__ Cf,
                              __bf16* __restrict__ Cbf,
                              float* __restrict__ H2,
                              const float* __restrict__ scale_p,
                              int M) {
    constexpr int K  = KC * 32;
    constexpr int Nn = NT * 16;
    const int wave = blockIdx.x * (blockDim.x >> 5) + (threadIdx.x >> 5);
    const int lane = threadIdx.x & 31;

    // --- stage packed B into LDS (whole block, before any wave exits) ---
    __shared__ __bf16 ldsb[USE_LDS ? (KC * NT * 512) : 32];
    if constexpr (USE_LDS) {
        constexpr int CH16 = KC * NT * 64;            // # of 16B chunks
        const v8bf* gsrc = (const v8bf*)Bpack;
        v8bf*       ldst = (v8bf*)ldsb;
        for (int i = threadIdx.x; i < CH16; i += blockDim.x) {
#if __has_builtin(__builtin_amdgcn_global_load_async_to_lds_b128)
            __builtin_amdgcn_global_load_async_to_lds_b128(
                (AS1 v4i*)(gsrc + i), (AS3 v4i*)(ldst + i), 0, 0);
#else
            ldst[i] = gsrc[i];                        // sync fallback
#endif
        }
#if __has_builtin(__builtin_amdgcn_global_load_async_to_lds_b128)
#if __has_builtin(__builtin_amdgcn_s_wait_asynccnt)
        __builtin_amdgcn_s_wait_asynccnt(0);
#else
        asm volatile("s_wait_asynccnt 0" ::: "memory");
#endif
#endif
        __syncthreads();
    }

    if (wave >= (M >> 4)) return;                     // wave-uniform: EXEC all-1s
    const int m0   = wave << 4;
    const int kgrp = (lane >> 4) << 3;
    const int col0 = lane & 15;

    v8f acc[NT] = {};

    const __bf16* arow  = A + (size_t)(m0 + (lane & 15)) * K;
    const __bf16* bbase = USE_LDS ? (const __bf16*)ldsb : Bpack;
    const __bf16* blane = bbase + (size_t)lane * 16;

#pragma unroll 2
    for (int kc32 = 0; kc32 < KC; ++kc32) {
        const int kc = kc32 * 32;
        if (kc + 32 < K)
            __builtin_prefetch(arow + kc + 32, 0, 1);

        const v8bf alo = *(const v8bf*)(arow + kc + kgrp);       // K=kgrp+0..7
        const v8bf ahi = *(const v8bf*)(arow + kc + 16 + kgrp);  // K=kgrp+16..23
        const v16bf afrag = __builtin_shufflevector(
            alo, ahi, 0, 1, 2, 3, 4, 5, 6, 7, 8, 9, 10, 11, 12, 13, 14, 15);

#pragma unroll
        for (int t = 0; t < NT; ++t) {
            const __bf16* bp = blane + (size_t)(kc32 * NT + t) * 32 * 16;
            const v8bf blo = *(const v8bf*)(bp);
            const v8bf bhi = *(const v8bf*)(bp + 8);
            const v16bf bfrag = __builtin_shufflevector(
                blo, bhi, 0, 1, 2, 3, 4, 5, 6, 7, 8, 9, 10, 11, 12, 13, 14, 15);
            acc[t] = __builtin_amdgcn_wmma_f32_16x16x32_bf16(
                false, afrag, false, bfrag, (short)0, acc[t], false, false);
        }
    }

    const int   rbase = (lane >> 4) << 3;             // D: VGPR r -> row rbase+r
    const float scl   = scale_p ? *scale_p : 0.0f;
#pragma unroll
    for (int t = 0; t < NT; ++t) {
        const float bval = bias[t * 16 + col0];
#pragma unroll
        for (int r = 0; r < 8; ++r) {
            const float  v   = acc[t][r] + bval;
            const size_t idx = (size_t)(m0 + rbase + r) * Nn + t * 16 + col0;
            if (Cf)  Cf[idx]  = v;
            if (Cbf) Cbf[idx] = f2bf(v);
            if (H2)  H2[idx]  = scl * v;
        }
    }
}

__global__ void gpr_zero_f32(float* __restrict__ p, size_t n) {
    size_t i = (size_t)blockIdx.x * blockDim.x + threadIdx.x;
    if (i < n) p[i] = 0.0f;
}

// One wave per edge: lanes cover H=128 features as float4 chunks.
// src/dst/w wave-uniform -> scalar loads; fp32 atomics resolve in L2.
__global__ void gpr_edge_scatter(const float* __restrict__ m,
                                 const float* __restrict__ w,
                                 const int* __restrict__ src,
                                 const int* __restrict__ dst,
                                 float* __restrict__ agg,
                                 int E, int Hdim) {
    const int gid  = blockIdx.x * blockDim.x + threadIdx.x;
    const int lane = gid & 31;
    const int e    = gid >> 5;
    if (e >= E) return;
    const int   s  = src[e];
    const int   d  = dst[e];
    const float we = w[e];
    const float4 mv = *(const float4*)(m + (size_t)s * Hdim + lane * 4);
    float* out = agg + (size_t)d * Hdim + lane * 4;
    __hip_atomic_fetch_add(out + 0, we * mv.x, __ATOMIC_RELAXED, __HIP_MEMORY_SCOPE_AGENT);
    __hip_atomic_fetch_add(out + 1, we * mv.y, __ATOMIC_RELAXED, __HIP_MEMORY_SCOPE_AGENT);
    __hip_atomic_fetch_add(out + 2, we * mv.z, __ATOMIC_RELAXED, __HIP_MEMORY_SCOPE_AGENT);
    __hip_atomic_fetch_add(out + 3, we * mv.w, __ATOMIC_RELAXED, __HIP_MEMORY_SCOPE_AGENT);
}

// xcur_bf = bf16(relu(agg)); hidden += temp_k * relu(agg)   (fused)
__global__ void gpr_relu_acc(const float* __restrict__ agg,
                             __bf16* __restrict__ xcur_bf,
                             float* __restrict__ hidden,
                             const float* __restrict__ tptr,
                             size_t n) {
    size_t i = (size_t)blockIdx.x * blockDim.x + threadIdx.x;
    if (i >= n) return;
    const float t = *tptr;
    float v = agg[i];
    v = v > 0.0f ? v : 0.0f;
    xcur_bf[i] = f2bf(v);
    hidden[i] += t * v;
}

extern "C" void kernel_launch(void* const* d_in, const int* in_sizes, int n_in,
                              void* d_out, int out_size, void* d_ws, size_t ws_size,
                              hipStream_t stream) {
    (void)in_sizes; (void)n_in; (void)out_size; (void)ws_size;

    const float* x     = (const float*)d_in[0];
    const float* w     = (const float*)d_in[1];
    const float* W_in  = (const float*)d_in[2];
    const float* b_in  = (const float*)d_in[3];
    const float* Wl    = (const float*)d_in[4];
    const float* bl    = (const float*)d_in[5];
    const float* temp  = (const float*)d_in[6];
    const float* W_out = (const float*)d_in[7];
    const float* b_out = (const float*)d_in[8];
    const int*   src   = (const int*)d_in[9];
    const int*   dst   = (const int*)d_in[10];
    float*       out   = (float*)d_out;

    constexpr int N = 50000, E = 800000, IN = 512, H = 128, OUT = 64, L = 4;
    const size_t nh = (size_t)N * H;

    // ---- workspace carve (256B-aligned regions) ----
    size_t off = 0;
    auto carve = [&](size_t bytes) -> char* {
        char* p = (char*)d_ws + off;
        off += (bytes + 255) & ~(size_t)255;
        return p;
    };
    float*  mbuf     = (float*) carve(nh * 4);          // projected features (f32)
    float*  agg      = (float*) carve(nh * 4);          // scatter accumulator
    float*  hidden   = (float*) carve(nh * 4);          // GPR accumulation
    __bf16* xbf      = (__bf16*)carve((size_t)N * IN * 2);
    __bf16* xcur_bf  = (__bf16*)carve(nh * 2);
    __bf16* hid_bf   = (__bf16*)carve(nh * 2);
    constexpr int NF_IN  = (IN / 32) * (H / 16);        // 128 frags (128 KB)
    constexpr int NF_L   = (H  / 32) * (H / 16);        // 32 frags  (32 KB)
    constexpr int NF_OUT = (H  / 32) * (OUT / 16);      // 16 frags  (16 KB)
    __bf16* bp_in  = (__bf16*)carve((size_t)NF_IN  * 512 * 2);
    __bf16* bp_l   = (__bf16*)carve((size_t)L * NF_L * 512 * 2);
    __bf16* bp_out = (__bf16*)carve((size_t)NF_OUT * 512 * 2);

    constexpr int WPB = 8;                              // 8 wave32 / 256-thr block
    const int mtiles = N / 16;                          // 3125 exact
    const dim3 gblk(WPB * 32);
    const dim3 ggrd((mtiles + WPB - 1) / WPB);
    const int eblocks = (int)(((size_t)E * 32 + 255) / 256);
    const int nblocks = (int)((nh + 255) / 256);

    // ---- one-time preprocessing ----
    const size_t xin4 = (size_t)N * IN / 4;
    gpr_f32_to_bf16<<<(int)((xin4 + 255) / 256), 256, 0, stream>>>(x, xbf, xin4);
    gpr_pack_b<<<(NF_IN * 32 + 255) / 256, 256, 0, stream>>>(W_in, bp_in, H, H / 16, NF_IN);
    for (int i = 0; i < L; ++i)
        gpr_pack_b<<<(NF_L * 32 + 255) / 256, 256, 0, stream>>>(
            Wl + (size_t)i * H * H, bp_l + (size_t)i * NF_L * 512, H, H / 16, NF_L);
    gpr_pack_b<<<(NF_OUT * 32 + 255) / 256, 256, 0, stream>>>(W_out, bp_out, OUT, OUT / 16, NF_OUT);

    // ---- h = x @ W_in + b_in ; xcur = bf16(h) ; hidden = temp[0]*h ----
    // B = 128 KB -> keep global path (stays L2-resident).
    gpr_gemm_wmma<8, 16, false><<<ggrd, gblk, 0, stream>>>(
        xbf, bp_in, b_in, (float*)nullptr, xcur_bf, hidden, temp + 0, N);

    for (int i = 0; i < L; ++i) {
        gpr_zero_f32<<<nblocks, 256, 0, stream>>>(agg, nh);
        // m = xcur @ Wl[i] + bl[i]   (B staged in 32 KB LDS, async copy)
        gpr_gemm_wmma<8, 4, true><<<ggrd, gblk, 0, stream>>>(
            xcur_bf, bp_l + (size_t)i * NF_L * 512, bl + (size_t)i * H,
            mbuf, (__bf16*)nullptr, (float*)nullptr, (const float*)nullptr, N);
        // agg[dst] += w * m[src]
        gpr_edge_scatter<<<eblocks, 256, 0, stream>>>(mbuf, w, src, dst, agg, E, H);
        // xcur = bf16(relu(agg)); hidden += temp[i+1]*relu(agg)
        gpr_relu_acc<<<nblocks, 256, 0, stream>>>(agg, xcur_bf, hidden, temp + (i + 1), nh);
    }

    // ---- out = hidden @ W_out + b_out   (B staged in 16 KB LDS) ----
    const size_t nh4 = nh / 4;
    gpr_f32_to_bf16<<<(int)((nh4 + 255) / 256), 256, 0, stream>>>(hidden, hid_bf, nh4);
    gpr_gemm_wmma<4, 4, true><<<ggrd, gblk, 0, stream>>>(
        hid_bf, bp_out, b_out, out, (__bf16*)nullptr, (float*)nullptr,
        (const float*)nullptr, N);
}
